// ConformerConvolutionQuant_82669530513996
// MI455X (gfx1250) — 8-bit matrix operand order repaired and checked by reading, compiled, not run on hardware
//
#include <hip/hip_runtime.h>
#include <math.h>


#define BB 32
#define TT 1500
#define FF 512
#define GG 1024
#define KW 31
#define MM (BB * TT)
#define EPSF 1e-5f

typedef __attribute__((ext_vector_type(8))) int v8i;


__device__ __forceinline__ float get_scale(const float* slot) {
  float m = *slot;
  float s = m * (1.0f / 127.0f);
  return s > 1e-8f ? s : 1e-8f;
}
__device__ __forceinline__ float fq_val(float x, float s) {
  float q = rintf(x / s);
  q = fminf(fmaxf(q, -128.0f), 127.0f);
  return q * s;
}
__device__ __forceinline__ signed char fq_int(float x, float s) {
  float q = rintf(x / s);
  q = fminf(fmaxf(q, -128.0f), 127.0f);
  return (signed char)(int)q;
}
__device__ __forceinline__ float sigmoidf(float y) { return 1.0f / (1.0f + expf(-y)); }

__device__ __forceinline__ void amax_block_reduce(float v, float* slot) {
  __shared__ float red[256];
  red[threadIdx.x] = v;
  __syncthreads();
  for (int s = 128; s > 0; s >>= 1) {
    if ((int)threadIdx.x < s) red[threadIdx.x] = fmaxf(red[threadIdx.x], red[threadIdx.x + s]);
    __syncthreads();
  }
  if (threadIdx.x == 0) atomicMax((unsigned int*)slot, __float_as_uint(red[0]));
}

__device__ __forceinline__ long a_pack_idx(int m, int k, int Kd) {
  int mt = m >> 4, r = m & 15;
  int kt = k >> 6, kk = k & 63;
  int lane = r + (((kk >> 3) & 1) << 4);
  int v = ((((kk >> 5) & 1) << 2) | (((kk >> 4) & 1) << 1) | ((kk >> 2) & 1));
  return (((long)(mt * (Kd >> 6) + kt)) << 10) + lane * 32 + v * 4 + (kk & 3);
}
__device__ __forceinline__ long b_pack_idx(int n, int k, int Kd) {
  int nt = n >> 4, r = n & 15;
  int kt = k >> 6, kk = k & 63;
  int lane = r + (((kk >> 3) & 1) << 4);
  int v = ((((kk >> 5) & 1) << 2) | (((kk >> 4) & 1) << 1) | ((kk >> 2) & 1));
  return (((long)(nt * (Kd >> 6) + kt)) << 10) + lane * 32 + v * 4 + (kk & 3);
}


__global__ void k_zero(float* p, int n) {
  int i = blockIdx.x * blockDim.x + threadIdx.x;
  if (i < n) p[i] = 0.0f;
}

__global__ void k_absmax(const float* __restrict__ x, long n, float* slot) {
  long i = blockIdx.x * (long)blockDim.x + threadIdx.x;
  long stride = (long)gridDim.x * blockDim.x;
  float m = 0.0f;
  for (; i < n; i += stride) m = fmaxf(m, fabsf(x[i]));
  amax_block_reduce(m, slot);
}

__global__ void k_quant(const float* __restrict__ x, signed char* __restrict__ q,
                        long n, const float* slot) {
  float s = get_scale(slot);
  long i = blockIdx.x * (long)blockDim.x + threadIdx.x;
  long stride = (long)gridDim.x * blockDim.x;
  for (; i < n; i += stride) q[i] = fq_int(x[i], s);
}

__global__ void k_quantB(const float* __restrict__ w, signed char* __restrict__ q,
                         long n, int Kd, const float* slot) {
  float s = get_scale(slot);
  long i = blockIdx.x * (long)blockDim.x + threadIdx.x;
  long stride = (long)gridDim.x * blockDim.x;
  for (; i < n; i += stride) {
    int nn = (int)(i >> 9);
    int kk = (int)(i & (FF - 1));
    q[b_pack_idx(nn, kk, Kd)] = fq_int(w[i], s);
  }
}

__global__ void k_fq_inplace(float* __restrict__ x, long n, const float* slot) {
  float s = get_scale(slot);
  long i = blockIdx.x * (long)blockDim.x + threadIdx.x;
  long stride = (long)gridDim.x * blockDim.x;
  for (; i < n; i += stride) x[i] = fq_val(x[i], s);
}

__global__ void k_ln(const float* __restrict__ x, const float* __restrict__ g,
                     const float* __restrict__ bta, signed char* __restrict__ qout,
                     float* slot, int phase) {
  const long row = blockIdx.x;
  const int tid = threadIdx.x;
  const float* xr = x + row * (long)FF;
  float v0 = xr[tid], v1 = xr[tid + 256];
  __shared__ float s1[256], s2[256];
  s1[tid] = v0 + v1;
  s2[tid] = v0 * v0 + v1 * v1;
  __syncthreads();
  for (int s = 128; s > 0; s >>= 1) {
    if (tid < s) { s1[tid] += s1[tid + s]; s2[tid] += s2[tid + s]; }
    __syncthreads();
  }
  float mu = s1[0] * (1.0f / FF);
  float var = s2[0] * (1.0f / FF) - mu * mu;
  float r = rsqrtf(var + EPSF);
  float o0 = (v0 - mu) * r * g[tid] + bta[tid];
  float o1 = (v1 - mu) * r * g[tid + 256] + bta[tid + 256];
  if (phase == 0) {
    amax_block_reduce(fmaxf(fabsf(o0), fabsf(o1)), slot);
  } else {
    float s = get_scale(slot);
    qout[a_pack_idx((int)row, tid, FF)] = fq_int(o0, s);
    qout[a_pack_idx((int)row, tid + 256, FF)] = fq_int(o1, s);
  }
}

__global__ void k_gemm_iu8(const signed char* __restrict__ Apk,
                           const signed char* __restrict__ Bpk,
                           const float* __restrict__ bias,
                           float* __restrict__ out,
                           int M, int N, int Kd,
                           const float* sA_slot, const float* sB_slot,
                           float* amax_slot) {
  const int lane = threadIdx.x & 31;
  const int wid = blockIdx.x * (blockDim.x >> 5) + (threadIdx.x >> 5);
  const int nquads = N >> 6;
  const int mp = wid / nquads;
  const int nq = wid % nquads;
  const int lhalf = lane >> 4;
  const int l16 = lane & 15;
  const int ktiles = Kd >> 6;

  const signed char* A0 = Apk + (((long)((mp * 2 + 0) * ktiles)) << 10) + lane * 32;
  const signed char* A1 = Apk + (((long)((mp * 2 + 1) * ktiles)) << 10) + lane * 32;
  const signed char* Bp0 = Bpk + (((long)((nq * 4 + 0) * ktiles)) << 10) + lane * 32;
  const signed char* Bp1 = Bpk + (((long)((nq * 4 + 1) * ktiles)) << 10) + lane * 32;
  const signed char* Bp2 = Bpk + (((long)((nq * 4 + 2) * ktiles)) << 10) + lane * 32;
  const signed char* Bp3 = Bpk + (((long)((nq * 4 + 3) * ktiles)) << 10) + lane * 32;

  const v8i vzero = {0, 0, 0, 0, 0, 0, 0, 0};
  v8i acc0[4], acc1[4];
#pragma unroll
  for (int j = 0; j < 4; ++j) { acc0[j] = vzero; acc1[j] = vzero; }

  for (int kt = 0; kt < ktiles; ++kt) {
    const long o = (long)kt << 10;
    v8i a0 = *(const v8i*)(A0 + o);
    v8i a1 = *(const v8i*)(A1 + o);
    v8i b0 = *(const v8i*)(Bp0 + o);
    v8i b1 = *(const v8i*)(Bp1 + o);
    v8i b2 = *(const v8i*)(Bp2 + o);
    v8i b3 = *(const v8i*)(Bp3 + o);
    acc0[0] = __builtin_amdgcn_wmma_i32_16x16x64_iu8(true, a0, true, b0, acc0[0], false, false);
    acc1[0] = __builtin_amdgcn_wmma_i32_16x16x64_iu8(true, a1, true, b0, acc1[0], false, false);
    acc0[1] = __builtin_amdgcn_wmma_i32_16x16x64_iu8(true, a0, true, b1, acc0[1], false, false);
    acc1[1] = __builtin_amdgcn_wmma_i32_16x16x64_iu8(true, a1, true, b1, acc1[1], false, false);
    acc0[2] = __builtin_amdgcn_wmma_i32_16x16x64_iu8(true, a0, true, b2, acc0[2], false, false);
    acc1[2] = __builtin_amdgcn_wmma_i32_16x16x64_iu8(true, a1, true, b2, acc1[2], false, false);
    acc0[3] = __builtin_amdgcn_wmma_i32_16x16x64_iu8(true, a0, true, b3, acc0[3], false, false);
    acc1[3] = __builtin_amdgcn_wmma_i32_16x16x64_iu8(true, a1, true, b3, acc1[3], false, false);
  }

  const float sc = get_scale(sA_slot) * get_scale(sB_slot);
  float lmax = 0.0f;
#pragma unroll
  for (int j = 0; j < 4; ++j) {
    const int c = nq * 64 + j * 16 + l16;
    const float bc = bias[c];
#pragma unroll
    for (int r = 0; r < 8; ++r) {
      int row0 = mp * 32 + r + lhalf * 8;
      int row1 = row0 + 16;
      float v0 = (float)acc0[j][r] * sc + bc;
      float v1 = (float)acc1[j][r] * sc + bc;
      out[(long)row0 * N + c] = v0;
      out[(long)row1 * N + c] = v1;
      lmax = fmaxf(lmax, fmaxf(fabsf(v0), fabsf(v1)));
    }
  }
  amax_block_reduce(lmax, amax_slot);
}

__global__ void k_glu(const float* __restrict__ in, signed char* __restrict__ qout,
                      const float* slot_in, float* slot_out, int phase) {
  float s = get_scale(slot_in);
  float sq = (phase == 1) ? get_scale(slot_out) : 0.0f;
  long n = (long)MM * FF;
  long i = blockIdx.x * (long)blockDim.x + threadIdx.x;
  long stride = (long)gridDim.x * blockDim.x;
  float lmax = 0.0f;
  for (; i < n; i += stride) {
    long row = i >> 9;
    int f = (int)(i & (FF - 1));
    float a = fq_val(in[row * (long)GG + f], s);
    float g = fq_val(in[row * (long)GG + FF + f], s);
    float v = a * sigmoidf(g);
    if (phase == 0) {
      lmax = fmaxf(lmax, fabsf(v));
    } else {
      long b = row / TT;
      long t = row - b * TT;
      qout[b * ((long)FF * TT) + (long)f * TT + t] = fq_int(v, sq);
    }
  }
  if (phase == 0) amax_block_reduce(lmax, slot_out);
}

__global__ void k_dwconv(const signed char* __restrict__ in, const signed char* __restrict__ w,
                         const float* __restrict__ bias, float* __restrict__ out,
                         const float* slot_in, const float* slot_w, float* slot_out) {
  float sc = get_scale(slot_in) * get_scale(slot_w);
  long i = blockIdx.x * (long)blockDim.x + threadIdx.x;
  long b = i / ((long)FF * TT);
  long rem = i - b * ((long)FF * TT);
  int f = (int)(rem / TT);
  int t = (int)(rem - (long)f * TT);
  const signed char* row = in + b * ((long)FF * TT) + (long)f * TT;
  const signed char* wf = w + f * KW;
  int acc = 0;
#pragma unroll
  for (int k = 0; k < KW; ++k) {
    int tt = t + k - (KW - 1) / 2;
    if (tt >= 0 && tt < TT) acc += (int)row[tt] * (int)wf[k];
  }
  float v = (float)acc * sc + bias[f];
  out[i] = v;
  amax_block_reduce(fabsf(v), slot_out);
}

__global__ void k_bnstat(const float* __restrict__ x, const float* slot,
                         float* __restrict__ psum, float* __restrict__ psumsq,
                         float* __restrict__ pmin, float* __restrict__ pmax) {
  float s = get_scale(slot);
  const float* base = x + blockIdx.x * (long)TT;
  float sum = 0.0f, sq = 0.0f, mn = 3.4e38f, mx = -3.4e38f;
  for (int t = threadIdx.x; t < TT; t += blockDim.x) {
    float v = fq_val(base[t], s);
    sum += v; sq += v * v;
    mn = fminf(mn, v); mx = fmaxf(mx, v);
  }
  __shared__ float s1[256], s2[256], s3[256], s4[256];
  s1[threadIdx.x] = sum; s2[threadIdx.x] = sq; s3[threadIdx.x] = mn; s4[threadIdx.x] = mx;
  __syncthreads();
  for (int st = 128; st > 0; st >>= 1) {
    if ((int)threadIdx.x < st) {
      s1[threadIdx.x] += s1[threadIdx.x + st];
      s2[threadIdx.x] += s2[threadIdx.x + st];
      s3[threadIdx.x] = fminf(s3[threadIdx.x], s3[threadIdx.x + st]);
      s4[threadIdx.x] = fmaxf(s4[threadIdx.x], s4[threadIdx.x + st]);
    }
    __syncthreads();
  }
  if (threadIdx.x == 0) {
    psum[blockIdx.x] = s1[0]; psumsq[blockIdx.x] = s2[0];
    pmin[blockIdx.x] = s3[0]; pmax[blockIdx.x] = s4[0];
  }
}

__global__ void k_bnfin(const float* __restrict__ psum, const float* __restrict__ psumsq,
                        const float* __restrict__ pmin, const float* __restrict__ pmax,
                        const float* __restrict__ g, const float* __restrict__ b,
                        float* __restrict__ scale, float* __restrict__ shift,
                        float* silu_amax_slot) {
  int f = blockIdx.x * blockDim.x + threadIdx.x;
  float cand = 0.0f;
  if (f < FF) {
    float sum = 0.0f, sq = 0.0f, mn = 3.4e38f, mx = -3.4e38f;
    for (int bb = 0; bb < BB; ++bb) {
      sum += psum[bb * FF + f];
      sq += psumsq[bb * FF + f];
      mn = fminf(mn, pmin[bb * FF + f]);
      mx = fmaxf(mx, pmax[bb * FF + f]);
    }
    float inv = 1.0f / ((float)BB * TT);
    float mu = sum * inv;
    float var = sq * inv - mu * mu;
    float sc = g[f] * rsqrtf(var + EPSF);
    float sh = b[f] - mu * sc;
    scale[f] = sc;
    shift[f] = sh;
    float ya = mn * sc + sh, yb = mx * sc + sh;
    float ylo = fminf(ya, yb), yhi = fmaxf(ya, yb);
    float slo = ylo * sigmoidf(ylo), shi = yhi * sigmoidf(yhi);
    cand = fmaxf(fabsf(slo), fabsf(shi));
    if (ylo <= -1.2784645f && -1.2784645f <= yhi) cand = fmaxf(cand, 0.2784646f);
  }
  amax_block_reduce(cand, silu_amax_slot);
}

__global__ void k_bnsilu_quant(const float* __restrict__ x, const float* slot_in,
                               const float* __restrict__ scale, const float* __restrict__ shift,
                               signed char* __restrict__ qout, const float* slot_q) {
  float s = get_scale(slot_in);
  float sq = get_scale(slot_q);
  long i = blockIdx.x * (long)blockDim.x + threadIdx.x;
  long b = i / ((long)FF * TT);
  long rem = i - b * ((long)FF * TT);
  int f = (int)(rem / TT);
  int t = (int)(rem - (long)f * TT);
  float v = fq_val(x[i], s);
  float y = v * scale[f] + shift[f];
  float sv = y * sigmoidf(y);
  int m = (int)(b * TT) + t;
  qout[a_pack_idx(m, f, FF)] = fq_int(sv, sq);
}


extern "C" void kernel_launch(void* const* d_in, const int* in_sizes, int n_in,
                              void* d_out, int out_size, void* d_ws, size_t ws_size,
                              hipStream_t stream) {
  const float* x    = (const float*)d_in[0];
  const float* ln_g = (const float*)d_in[1];
  const float* ln_b = (const float*)d_in[2];
  const float* W1   = (const float*)d_in[3];
  const float* b1   = (const float*)d_in[4];
  const float* dww  = (const float*)d_in[5];
  const float* dwb  = (const float*)d_in[6];
  const float* bn_g = (const float*)d_in[7];
  const float* bn_b = (const float*)d_in[8];
  const float* W2   = (const float*)d_in[9];
  const float* b2   = (const float*)d_in[10];
  float* out = (float*)d_out;

  char* ws = (char*)d_ws;
  size_t off = 0;
  auto take = [&](size_t bytes) -> char* {
    char* p = ws + off;
    off = (off + bytes + 255) & ~(size_t)255;
    return p;
  };
  float* scal = (float*)take(16 * sizeof(float));
  float* bnsc = (float*)take(FF * sizeof(float));
  float* bnsh = (float*)take(FF * sizeof(float));
  float* psum = (float*)take((size_t)BB * FF * sizeof(float));
  float* psq  = (float*)take((size_t)BB * FF * sizeof(float));
  float* pmn  = (float*)take((size_t)BB * FF * sizeof(float));
  float* pmx  = (float*)take((size_t)BB * FF * sizeof(float));
  signed char* W1q = (signed char*)take((size_t)GG * FF);
  signed char* W2q = (signed char*)take((size_t)FF * FF);
  signed char* dwq = (signed char*)take((size_t)FF * KW);
  float* buf0 = (float*)take((size_t)MM * GG * sizeof(float));
  signed char* buf2 = (signed char*)take((size_t)MM * FF);
  signed char* buf3 = (signed char*)take((size_t)MM * FF);

  float* a_ln = scal + 0, *a_w1 = scal + 1, *a_g1 = scal + 2, *a_glu = scal + 3;
  float* a_dw = scal + 4, *a_cv = scal + 5, *a_si = scal + 6, *a_w2 = scal + 7, *a_g2 = scal + 8;

  const long nBTF = (long)MM * FF;

  k_zero<<<1, 32, 0, stream>>>(scal, 16);

  k_absmax<<<512, 256, 0, stream>>>(W1, (long)GG * FF, a_w1);
  k_absmax<<<62, 256, 0, stream>>>(dww, (long)FF * KW, a_dw);
  k_absmax<<<256, 256, 0, stream>>>(W2, (long)FF * FF, a_w2);
  k_quantB<<<512, 256, 0, stream>>>(W1, W1q, (long)GG * FF, FF, a_w1);
  k_quant<<<62, 256, 0, stream>>>(dww, dwq, (long)FF * KW, a_dw);
  k_quantB<<<256, 256, 0, stream>>>(W2, W2q, (long)FF * FF, FF, a_w2);

  k_ln<<<MM, 256, 0, stream>>>(x, ln_g, ln_b, nullptr, a_ln, 0);
  k_ln<<<MM, 256, 0, stream>>>(x, ln_g, ln_b, buf2, a_ln, 1);
  k_gemm_iu8<<<(MM / 32) * (GG / 64) / 8, 256, 0, stream>>>(buf2, W1q, b1, buf0,
                                                            MM, GG, FF, a_ln, a_w1, a_g1);
  k_glu<<<8192, 256, 0, stream>>>(buf0, nullptr, a_g1, a_glu, 0);
  k_glu<<<8192, 256, 0, stream>>>(buf0, buf3, a_g1, a_glu, 1);

  k_dwconv<<<(int)(nBTF / 256), 256, 0, stream>>>(buf3, dwq, dwb, buf0, a_glu, a_dw, a_cv);

  k_bnstat<<<BB * FF, 256, 0, stream>>>(buf0, a_cv, psum, psq, pmn, pmx);
  k_bnfin<<<2, 256, 0, stream>>>(psum, psq, pmn, pmx, bn_g, bn_b, bnsc, bnsh, a_si);
  k_bnsilu_quant<<<(int)(nBTF / 256), 256, 0, stream>>>(buf0, a_cv, bnsc, bnsh, buf2, a_si);

  k_gemm_iu8<<<(MM / 32) * (FF / 64) / 8, 256, 0, stream>>>(buf2, W2q, b2, out,
                                                            MM, FF, FF, a_si, a_w2, a_g2);
  k_fq_inplace<<<8192, 256, 0, stream>>>(out, nBTF, a_g2);
}
